// GCN_pyg_72353019068829
// MI455X (gfx1250) — matrix operand order repaired and checked by reading, not compiled, not run on hardware
//
#include <hip/hip_runtime.h>
#include <stdint.h>

#define NN 100000          // nodes
#define EE 1200000         // edges
#define DD 64              // emb dim
#define GG 1024            // graphs
#define TILES (NN / 16)    // 6250 exact
#define ET (EE + NN)       // edges incl. self-loops

typedef __attribute__((ext_vector_type(16))) __bf16 v16bf;
typedef __attribute__((ext_vector_type(8)))  float  v8f;

static __device__ __forceinline__ unsigned short f2bf(float f) {
  union { float f; unsigned int u; } x; x.f = f;
  unsigned int u = x.u;
  unsigned int r = (u + 0x7FFFu + ((u >> 16) & 1u)) >> 16;   // round-to-nearest-even
  return (unsigned short)r;
}

// ---------------- AtomEncoder: h[n,d] = sum of 3 embedding gathers ----------------
__global__ void k_atom_encoder(const int* __restrict__ x_atom,
                               const float* __restrict__ atom_emb,
                               float* __restrict__ h) {
  int gid = blockIdx.x * blockDim.x + threadIdx.x;
  if (gid >= NN * DD) return;
  int n = gid >> 6, d = gid & 63;
  int i0 = x_atom[n * 3 + 0], i1 = x_atom[n * 3 + 1], i2 = x_atom[n * 3 + 2];
  const float* e0 = atom_emb;
  const float* e1 = atom_emb + 128 * DD;
  const float* e2 = atom_emb + 2 * 128 * DD;
  h[gid] = e0[i0 * DD + d] + e1[i1 * DD + d] + e2[i2 * DD + d];
}

// ---- Pack h (fp32 row-major) -> bf16 WMMA A-layout (16-bit A 16x32, ISA 7.12.2) ----
__global__ void k_pack_h(const float* __restrict__ h, unsigned short* __restrict__ hpack) {
  int gid = blockIdx.x * blockDim.x + threadIdx.x;
  if (gid >= TILES * 2 * 32) return;
  int lane = gid & 31;
  int kb   = (gid >> 5) & 1;
  int tile = gid >> 6;
  int m  = tile * 16 + (lane & 15);
  int hi = lane >> 4;
  unsigned short tmp[16];
#pragma unroll
  for (int e = 0; e < 16; ++e) {
    int k = kb * 32 + ((e >> 3) << 4) + (hi << 3) + (e & 7);
    tmp[e] = f2bf(h[(size_t)m * DD + k]);
  }
  uint4* dst = (uint4*)(hpack + (size_t)gid * 16);
  dst[0] = *(const uint4*)&tmp[0];
  dst[1] = *(const uint4*)&tmp[8];
}

// ---- Pack conv_W (9 x 64x64 row-major, W[k][n]) -> bf16 WMMA B-layout (32x16) ----
__global__ void k_pack_w(const float* __restrict__ W, unsigned short* __restrict__ wpack) {
  int gid = blockIdx.x * blockDim.x + threadIdx.x;
  if (gid >= 9 * 8 * 32) return;
  int lane = gid & 31;
  int sub  = (gid >> 5) & 7;      // kb*4 + nt
  int mat  = gid >> 8;
  int kb = sub >> 2, nt = sub & 3;
  int hi = lane >> 4, n = nt * 16 + (lane & 15);
  unsigned short tmp[16];
#pragma unroll
  for (int e = 0; e < 16; ++e) {
    int k = kb * 32 + ((e >> 3) << 4) + (hi << 3) + (e & 7);
    tmp[e] = f2bf(W[((size_t)mat * 64 + k) * 64 + n]);
  }
  uint4* dst = (uint4*)(wpack + (size_t)gid * 16);
  dst[0] = *(const uint4*)&tmp[0];
  dst[1] = *(const uint4*)&tmp[8];
}

// ---------------- Degrees (shared across layers): deg_k[i] = 1 + sum_{dst==i} w ----
__global__ void k_deg_init(float* __restrict__ deg3) {
  int gid = blockIdx.x * blockDim.x + threadIdx.x;
  if (gid < 3 * NN) deg3[gid] = 1.0f;   // self-loop weight
}
__global__ void k_deg_accum(const int* __restrict__ edge_index,
                            const float* __restrict__ edge_attr,
                            float* __restrict__ deg3) {
  int e = blockIdx.x * blockDim.x + threadIdx.x;
  if (e >= EE) return;
  int d = edge_index[EE + e];
  atomicAdd(&deg3[0 * NN + d], edge_attr[e * 3 + 0]);
  atomicAdd(&deg3[1 * NN + d], edge_attr[e * 3 + 1]);
  atomicAdd(&deg3[2 * NN + d], edge_attr[e * 3 + 2]);
}
__global__ void k_dinv(float* __restrict__ deg3) {
  int gid = blockIdx.x * blockDim.x + threadIdx.x;
  if (gid >= 3 * NN) return;
  float v = deg3[gid];
  deg3[gid] = (v > 0.0f) ? rsqrtf(v) : 0.0f;
}

// ---- Edge norms, layer-invariant: norm3[k][e] = dinv_k[s]*w_k*dinv_k[t] ----------
__global__ void k_norm3(const int* __restrict__ edge_index,
                        const float* __restrict__ edge_attr,
                        const float* __restrict__ dinv3,
                        float* __restrict__ norm3) {
  int e = blockIdx.x * blockDim.x + threadIdx.x;
  if (e >= ET) return;
  int s, t; float w0, w1, w2;
  if (e < EE) {
    s = edge_index[e]; t = edge_index[EE + e];
    w0 = edge_attr[(size_t)e * 3 + 0];
    w1 = edge_attr[(size_t)e * 3 + 1];
    w2 = edge_attr[(size_t)e * 3 + 2];
  } else {
    s = t = e - EE; w0 = w1 = w2 = 1.0f;
  }
  norm3[0 * ET + e] = dinv3[0 * NN + s] * w0 * dinv3[0 * NN + t];
  norm3[1 * ET + e] = dinv3[1 * NN + s] * w1 * dinv3[1 * NN + t];
  norm3[2 * ET + e] = dinv3[2 * NN + s] * w2 * dinv3[2 * NN + t];
}

// ---------------- WMMA GEMM: xw = h(bf16, packed) @ W(bf16, packed) -> f32 -------
// One wave per 16-node tile; 2 K-blocks x 4 N-tiles = 8 v_wmma per wave.
__global__ void __launch_bounds__(64) k_gemm_wmma(const unsigned short* __restrict__ hpack,
                                                  const unsigned short* __restrict__ wpack_mat,
                                                  float* __restrict__ xw) {
  int lane = threadIdx.x & 31;
  int tile = blockIdx.x * 2 + (threadIdx.x >> 5);   // TILES even -> no tail, EXEC all-1s
  const v16bf* A = (const v16bf*)hpack + (size_t)tile * 64;
  const v16bf* B = (const v16bf*)wpack_mat;
  v8f c0 = {}, c1 = {}, c2 = {}, c3 = {};
#pragma unroll
  for (int kb = 0; kb < 2; ++kb) {
    v16bf a  = A[kb * 32 + lane];
    v16bf b0 = B[(kb * 4 + 0) * 32 + lane];
    v16bf b1 = B[(kb * 4 + 1) * 32 + lane];
    v16bf b2 = B[(kb * 4 + 2) * 32 + lane];
    v16bf b3 = B[(kb * 4 + 3) * 32 + lane];
    c0 = __builtin_amdgcn_wmma_f32_16x16x32_bf16(false, a, false, b0, (short)0, c0, false, false);
    c1 = __builtin_amdgcn_wmma_f32_16x16x32_bf16(false, a, false, b1, (short)0, c1, false, false);
    c2 = __builtin_amdgcn_wmma_f32_16x16x32_bf16(false, a, false, b2, (short)0, c2, false, false);
    c3 = __builtin_amdgcn_wmma_f32_16x16x32_bf16(false, a, false, b3, (short)0, c3, false, false);
  }
  // C/D layout: VGPR r holds (m = r + 8*(lane>>4), n = lane&15)
  int hi = lane >> 4, nlo = lane & 15;
  float* out = xw + (size_t)tile * 16 * DD;
#pragma unroll
  for (int r = 0; r < 8; ++r) {
    int m = r + 8 * hi;
    out[m * DD + 0 * 16 + nlo] = c0[r];
    out[m * DD + 1 * 16 + nlo] = c1[r];
    out[m * DD + 2 * 16 + nlo] = c2[r];
    out[m * DD + 3 * 16 + nlo] = c3[r];
  }
}

// ---------------- acc init with summed branch biases for this layer ----------------
__global__ void k_acc_init(float* __restrict__ acc, const float* __restrict__ b_layer) {
  int gid = blockIdx.x * blockDim.x + threadIdx.x;
  if (gid >= NN * DD) return;
  int d = gid & 63;
  acc[gid] = b_layer[d] + b_layer[64 + d] + b_layer[128 + d];
}

// ---- Fused scatter, all 3 branches: one b128 gather per branch, ONE atomic/elem ----
// 16 threads per edge, 4 floats each.
__global__ void k_scatter3(const int* __restrict__ edge_index,
                           const float* __restrict__ norm3,
                           const float* __restrict__ xw3,   // 3 x [NN][DD]
                           float* __restrict__ acc) {
  long gid = (long)blockIdx.x * blockDim.x + threadIdx.x;
  if (gid >= (long)ET * 16) return;
  int e = (int)(gid >> 4), q = (int)(gid & 15) * 4;
  int s, t;
  if (e < EE) { s = edge_index[e]; t = edge_index[EE + e]; }
  else        { s = t = e - EE; }
  float n0 = norm3[0 * ET + e];
  float n1 = norm3[1 * ET + e];
  float n2 = norm3[2 * ET + e];
  const float4 x0 = *(const float4*)(xw3 + 0 * (size_t)NN * DD + (size_t)s * DD + q);
  const float4 x1 = *(const float4*)(xw3 + 1 * (size_t)NN * DD + (size_t)s * DD + q);
  const float4 x2 = *(const float4*)(xw3 + 2 * (size_t)NN * DD + (size_t)s * DD + q);
  float v0 = x0.x * n0 + x1.x * n1 + x2.x * n2;
  float v1 = x0.y * n0 + x1.y * n1 + x2.y * n2;
  float v2 = x0.z * n0 + x1.z * n1 + x2.z * n2;
  float v3 = x0.w * n0 + x1.w * n1 + x2.w * n2;
  float* a = acc + (size_t)t * DD + q;
  atomicAdd(a + 0, v0);
  atomicAdd(a + 1, v1);
  atomicAdd(a + 2, v2);
  atomicAdd(a + 3, v3);
}

// ---- Fallback scatter, single branch (used if workspace too small for xw3) -------
__global__ void k_scatter1(const int* __restrict__ edge_index,
                           const float* __restrict__ norm,   // norm3 + k*ET
                           const float* __restrict__ xw,
                           float* __restrict__ acc) {
  long gid = (long)blockIdx.x * blockDim.x + threadIdx.x;
  if (gid >= (long)ET * 16) return;
  int e = (int)(gid >> 4), q = (int)(gid & 15) * 4;
  int s, t;
  if (e < EE) { s = edge_index[e]; t = edge_index[EE + e]; }
  else        { s = t = e - EE; }
  float nrm = norm[e];
  const float4 x = *(const float4*)(xw + (size_t)s * DD + q);
  float* a = acc + (size_t)t * DD + q;
  atomicAdd(a + 0, x.x * nrm);
  atomicAdd(a + 1, x.y * nrm);
  atomicAdd(a + 2, x.z * nrm);
  atomicAdd(a + 3, x.w * nrm);
}

// ---------------- residual + optional relu: h += relu?(acc) ----------------------
__global__ void k_update(float* __restrict__ h, const float* __restrict__ acc, int do_relu) {
  int gid = blockIdx.x * blockDim.x + threadIdx.x;
  if (gid >= NN * DD) return;
  float v = acc[gid];
  if (do_relu) v = fmaxf(v, 0.0f);
  h[gid] += v;
}

// ---------------- mean pool over graphs ------------------------------------------
__global__ void k_pool_zero(float* __restrict__ hg, float* __restrict__ cnt) {
  int gid = blockIdx.x * blockDim.x + threadIdx.x;
  if (gid < GG * DD) hg[gid] = 0.0f;
  if (gid < GG) cnt[gid] = 0.0f;
}
__global__ void k_pool_sum(const float* __restrict__ h, const int* __restrict__ batch,
                           float* __restrict__ hg, float* __restrict__ cnt) {
  int gid = blockIdx.x * blockDim.x + threadIdx.x;
  if (gid >= NN * 16) return;
  int n = gid >> 4, q = (gid & 15) * 4;
  int g = batch[n];
  const float4 x = *(const float4*)(h + (size_t)n * DD + q);
  float* a = hg + (size_t)g * DD + q;
  atomicAdd(a + 0, x.x);
  atomicAdd(a + 1, x.y);
  atomicAdd(a + 2, x.z);
  atomicAdd(a + 3, x.w);
  if (q == 0) atomicAdd(&cnt[g], 1.0f);
}
__global__ void k_pool_div(float* __restrict__ hg, const float* __restrict__ cnt) {
  int gid = blockIdx.x * blockDim.x + threadIdx.x;
  if (gid >= GG * DD) return;
  hg[gid] /= fmaxf(cnt[gid >> 6], 1.0f);
}

// ---------------- tiny FC stack ---------------------------------------------------
__global__ void k_fc(const float* __restrict__ in, const float* __restrict__ W,
                     const float* __restrict__ b, float* __restrict__ out) {
  int gid = blockIdx.x * blockDim.x + threadIdx.x;
  if (gid >= GG * DD) return;
  int g = gid >> 6, o = gid & 63;
  float s = b[o];
#pragma unroll 8
  for (int kk = 0; kk < DD; ++kk) s += in[(size_t)g * DD + kk] * W[kk * DD + o];
  out[gid] = s;
}
__global__ void k_fc3(const float* __restrict__ in, const float* __restrict__ W3,
                      const float* __restrict__ b3, float* __restrict__ out) {
  int g = blockIdx.x * blockDim.x + threadIdx.x;
  if (g >= GG) return;
  float s = b3[0];
#pragma unroll 8
  for (int kk = 0; kk < DD; ++kk) s += in[(size_t)g * DD + kk] * W3[kk];
  out[g] = s;
}

extern "C" void kernel_launch(void* const* d_in, const int* in_sizes, int n_in,
                              void* d_out, int out_size, void* d_ws, size_t ws_size,
                              hipStream_t stream) {
  (void)in_sizes; (void)n_in; (void)out_size;
  const int*   x_atom     = (const int*)d_in[0];
  const int*   edge_index = (const int*)d_in[1];
  const float* edge_attr  = (const float*)d_in[2];
  const int*   batch      = (const int*)d_in[3];
  const float* atom_emb   = (const float*)d_in[4];
  const float* conv_W     = (const float*)d_in[5];
  const float* conv_b     = (const float*)d_in[6];
  const float* fc_W1 = (const float*)d_in[7];
  const float* fc_b1 = (const float*)d_in[8];
  const float* fc_W2 = (const float*)d_in[9];
  const float* fc_b2 = (const float*)d_in[10];
  const float* fc_W3 = (const float*)d_in[11];
  const float* fc_b3 = (const float*)d_in[12];
  float* out = (float*)d_out;

  // ---- workspace carve-up ----
  char* ws = (char*)d_ws; size_t off = 0;
  auto carve = [&](size_t bytes) -> void* {
    void* p = ws + off; off += (bytes + 255) & ~(size_t)255; return p;
  };
  const size_t ndB = (size_t)NN * DD * 4;
  float*          h     = (float*)carve(ndB);
  float*          acc   = (float*)carve(ndB);
  unsigned short* hpack = (unsigned short*)carve((size_t)TILES * 2 * 32 * 16 * 2);
  unsigned short* wpack = (unsigned short*)carve((size_t)9 * 8 * 32 * 16 * 2);
  float*          deg3  = (float*)carve((size_t)3 * NN * 4);
  float*          norm3 = (float*)carve((size_t)3 * ET * 4);
  float*          hg    = (float*)carve((size_t)GG * DD * 4);
  float*          cnt   = (float*)carve((size_t)GG * 4);
  float*          t1    = (float*)carve((size_t)GG * DD * 4);
  float*          t2    = (float*)carve((size_t)GG * DD * 4);
  // fused path keeps all 3 branch GEMM outputs live (one atomic per element per layer)
  bool fused = (ws_size >= off + 3 * (ndB + 256));
  float* xw = (float*)carve(fused ? 3 * ndB : ndB);

  const int B = 256;
  const int nd_blk  = (NN * DD + B - 1) / B;
  const int n3_blk  = (3 * NN + B - 1) / B;
  const int et_blk  = (ET + B - 1) / B;
  const int sc_blk  = (int)(((long)ET * 16 + B - 1) / B);

  // node features + weight packing + degree/norm precompute (layer-invariant)
  k_atom_encoder<<<nd_blk, B, 0, stream>>>(x_atom, atom_emb, h);
  k_pack_w<<<(9 * 8 * 32 + B - 1) / B, B, 0, stream>>>(conv_W, wpack);
  k_deg_init<<<n3_blk, B, 0, stream>>>(deg3);
  k_deg_accum<<<(EE + B - 1) / B, B, 0, stream>>>(edge_index, edge_attr, deg3);
  k_dinv<<<n3_blk, B, 0, stream>>>(deg3);
  k_norm3<<<et_blk, B, 0, stream>>>(edge_index, edge_attr, deg3, norm3);

  for (int layer = 0; layer < 3; ++layer) {
    k_pack_h<<<(TILES * 2 * 32 + B - 1) / B, B, 0, stream>>>(h, hpack);
    k_acc_init<<<nd_blk, B, 0, stream>>>(acc, conv_b + (size_t)layer * 3 * DD);
    if (fused) {
      for (int k = 0; k < 3; ++k) {
        const unsigned short* wmat = wpack + ((size_t)(layer * 3 + k) * 8) * 32 * 16;
        k_gemm_wmma<<<TILES / 2, 64, 0, stream>>>(hpack, wmat, xw + (size_t)k * NN * DD);
      }
      k_scatter3<<<sc_blk, B, 0, stream>>>(edge_index, norm3, xw, acc);
    } else {
      for (int k = 0; k < 3; ++k) {
        const unsigned short* wmat = wpack + ((size_t)(layer * 3 + k) * 8) * 32 * 16;
        k_gemm_wmma<<<TILES / 2, 64, 0, stream>>>(hpack, wmat, xw);
        k_scatter1<<<sc_blk, B, 0, stream>>>(edge_index, norm3 + (size_t)k * ET, xw, acc);
      }
    }
    k_update<<<nd_blk, B, 0, stream>>>(h, acc, layer < 2 ? 1 : 0);
  }

  // mean pool + FC head
  k_pool_zero<<<(GG * DD + B - 1) / B, B, 0, stream>>>(hg, cnt);
  k_pool_sum<<<(NN * 16 + B - 1) / B, B, 0, stream>>>(h, batch, hg, cnt);
  k_pool_div<<<(GG * DD + B - 1) / B, B, 0, stream>>>(hg, cnt);
  k_fc <<<(GG * DD + B - 1) / B, B, 0, stream>>>(hg, fc_W1, fc_b1, t1);
  k_fc <<<(GG * DD + B - 1) / B, B, 0, stream>>>(t1, fc_W2, fc_b2, t2);
  k_fc3<<<(GG + B - 1) / B, B, 0, stream>>>(t2, fc_W3, fc_b3, out);
}
